// AnchorSet2NodeMPNN_56023553409076
// MI455X (gfx1250) — compile-verified, matrix operand order checked by reading, not run on hardware
//
#include <hip/hip_runtime.h>
#include <hip/hip_bf16.h>

// ---------------------------------------------------------------------------
// AnchorSet2NodeMPNN on MI455X (gfx1250): bf16 WMMA pipeline.
// B=8, N=384, A=64, H=256, E=64.  Layer-1 of the pair MLP is decomposed:
//   z1 = rbf @ W1a_rbf  +  (nf @ W1a_n + b1a)  +  (af @ W1a_a)
// so per-pair GEMM work drops from K=576 to K=64 for layer 1.
// K-loops are fully-unrolled ping-pong pipelines over 16x64 strips with 4
// accumulators; amdgpu_waves_per_eu(1) gives the register allocator room to
// keep both fragment buffers in flight instead of serializing on loadcnt 0.
// ---------------------------------------------------------------------------

typedef __attribute__((ext_vector_type(16))) __bf16 v16bf;
typedef __attribute__((ext_vector_type(8)))  float  v8f;

struct Frag32 { uint4 lo, hi; };

__device__ __forceinline__ v8f vzero() {
  v8f z;
  #pragma unroll
  for (int i = 0; i < 8; ++i) z[i] = 0.0f;
  return z;
}

// Fragment loader for both A (16xK act rows, row-major [M][K]) and
// B (Kx16 weights stored as [N][K] row-major).  Per ISA 16-bit A layout:
// lanes 0-15 hold K={0..7,16..23}, lanes 16-31 hold K={8..15,24..31} of row
// (lane & 15).  stride must be a multiple of 8 elements (16B alignment).
__device__ __forceinline__ v16bf load_frag(const __bf16* p0, int stride, int lane) {
  const int r = lane & 15;
  const int h = lane >> 4;
  const __bf16* p = p0 + r * stride + h * 8;
  Frag32 f;
  f.lo = *(const uint4*)(p);
  f.hi = *(const uint4*)(p + 16);
  return __builtin_bit_cast(v16bf, f);
}

__device__ __forceinline__ v8f wmma_bf16(v16bf a, v16bf b, v8f c) {
  return __builtin_amdgcn_wmma_f32_16x16x32_bf16(false, a, false, b, (short)0, c,
                                                 false, false);
}

// Fully-unrolled, software-pipelined K-loop over a 16 x (NT*16) strip.
// Ping-pong fragment buffers: while WMMAs for step k issue, the loads for
// step k+1 are already in flight (partial s_wait_loadcnt instead of 0).
// KS must be even (all call sites use 2/8/16).
template <int NT, int KS>
__device__ __forceinline__ void wmma_strip(const __bf16* A, int sA,
                                           const __bf16* Bg, int sB,
                                           int lane, v8f* acc) {
  static_assert(KS % 2 == 0, "KS must be even");
  v16bf a0 = load_frag(A, sA, lane);
  v16bf b0[NT];
  #pragma unroll
  for (int t = 0; t < NT; ++t) b0[t] = load_frag(Bg + t * 16 * sB, sB, lane);
  #pragma unroll
  for (int ks = 0; ks < KS; ks += 2) {
    // Speculative prefetch ahead; OOB prefetches are dropped by HW.
    __builtin_prefetch((const void*)(Bg + (ks + 4) * 32 + (lane & 15) * sB), 0, 1);
    v16bf a1 = load_frag(A + (ks + 1) * 32, sA, lane);
    v16bf b1[NT];
    #pragma unroll
    for (int t = 0; t < NT; ++t)
      b1[t] = load_frag(Bg + (ks + 1) * 32 + t * 16 * sB, sB, lane);
    #pragma unroll
    for (int t = 0; t < NT; ++t) acc[t] = wmma_bf16(a0, b0[t], acc[t]);
    if (ks + 2 < KS) {
      a0 = load_frag(A + (ks + 2) * 32, sA, lane);
      #pragma unroll
      for (int t = 0; t < NT; ++t)
        b0[t] = load_frag(Bg + (ks + 2) * 32 + t * 16 * sB, sB, lane);
    }
    #pragma unroll
    for (int t = 0; t < NT; ++t) acc[t] = wmma_bf16(a1, b1[t], acc[t]);
  }
}

// ---------------------------------------------------------------------------
// Prep kernels
// ---------------------------------------------------------------------------
__global__ void k_convert_bf16(const float* __restrict__ in,
                               __bf16* __restrict__ out, int n) {
  int i = blockIdx.x * blockDim.x + threadIdx.x;
  if (i < n) out[i] = (__bf16)in[i];
}

// out[n*K + k] = W[(rowOff+k)*ldw + n]   (weight transpose + bf16 convert)
__global__ void k_transpose_convert(const float* __restrict__ W, int ldw,
                                    int rowOff, int K, int N,
                                    __bf16* __restrict__ out) {
  int idx = blockIdx.x * blockDim.x + threadIdx.x;
  if (idx >= N * K) return;
  int n = idx / K, k = idx - n * K;
  out[idx] = (__bf16)W[(rowOff + k) * ldw + n];
}

// C[M][N] f32 = A_bf16[M][256] @ B_bf16[N][256]^T (+bias).  K fixed at 256
// (8 WMMA K-steps).  8 waves/block, one 16x64 strip per wave; grid must
// cover (M/16)*(N/64) waves exactly.
__global__ __launch_bounds__(256)
__attribute__((amdgpu_waves_per_eu(1))) void k_gemm_bf16_k256(
    const __bf16* __restrict__ A, const __bf16* __restrict__ Bw,
    const float* __restrict__ bias, float* __restrict__ C, int M, int N) {
  const int K = 256;
  int lane = threadIdx.x & 31;
  int wid  = blockIdx.x * 8 + (threadIdx.x >> 5);
  int tilesN = N >> 6;
  int tm = wid / tilesN, tn = wid - tm * tilesN;
  if (tm * 16 >= M) return;                 // wave-uniform
  const __bf16* Ab = A + (size_t)tm * 16 * K;
  int c0 = tn * 64;
  v8f acc[4] = {vzero(), vzero(), vzero(), vzero()};
  wmma_strip<4, 8>(Ab, K, Bw + (size_t)c0 * K, K, lane, acc);
  #pragma unroll
  for (int t = 0; t < 4; ++t) {
    int col = c0 + t * 16 + (lane & 15);
    float bb = bias ? bias[col] : 0.0f;
    #pragma unroll
    for (int v = 0; v < 8; ++v) {
      int row = tm * 16 + v + (lane >> 4) * 8;
      C[(size_t)row * N + col] = acc[t][v] + bb;
    }
  }
}

// ---------------------------------------------------------------------------
// Main fused kernel: 1 block per 16-node tile, 256 threads (8 wave32).
// ---------------------------------------------------------------------------
__global__ __launch_bounds__(256)
__attribute__((amdgpu_waves_per_eu(1))) void k_mpnn_main(
    const float* __restrict__ node_x, const float* __restrict__ anchor_x,
    const float* __restrict__ node_features, const float* __restrict__ node_mask,
    const __bf16* __restrict__ w1an,  // [512][256]  (nf part of W1a, transposed)
    const float*  __restrict__ t_anchor, // [B*A][512] f32
    const __bf16* __restrict__ w1ar,  // [512][64]
    const __bf16* __restrict__ w1bt,  // [512][512]
    const __bf16* __restrict__ w1ct,  // [256][512]
    const __bf16* __restrict__ w2at,  // [512][256]
    const __bf16* __restrict__ w2bt,  // [512][512]
    const __bf16* __restrict__ w2ct,  // [256][512]
    const float* __restrict__ b1a, const float* __restrict__ b1b,
    const float* __restrict__ b1c,
    const float* __restrict__ ln1g, const float* __restrict__ ln1b,
    const float* __restrict__ b2a, const float* __restrict__ b2b,
    const float* __restrict__ b2c,
    const float* __restrict__ ln2g, const float* __restrict__ ln2b,
    float* __restrict__ out) {
  __shared__ float s_d[16];
  __shared__ __align__(16) __bf16 s_rbf[16 * 64];
  __shared__ __align__(16) __bf16 s_h1[16 * 512];
  __shared__ __align__(16) __bf16 s_h2[16 * 512];
  __shared__ __align__(16) unsigned char s_un[16 * 512 * 2]; // 16KB union
  __shared__ __align__(16) __bf16 s_nf1[16 * 256];           // nf bf16 / LN1 out

  __bf16* s_zn = (__bf16*)s_un;   // [16][512] layer-1 node term (anchor loop)
  float*  s_x  = (float*)s_un;    // [16][256] f32 scratch (after the loop)

  const int tid  = threadIdx.x;
  const int lane = tid & 31;
  const int w    = tid >> 5;                 // wave 0..7
  const int node0 = blockIdx.x * 16;
  const int g = node0 / 384;                 // graph index

  // ---- stage 0: node-feature tile -> bf16 LDS -----------------------------
  for (int i = tid; i < 16 * 256; i += 256) {
    int row = i >> 8, col = i & 255;
    s_nf1[i] = (__bf16)node_features[(size_t)(node0 + row) * 256 + col];
  }
  __syncthreads();

  // ---- stage 1: z_node = nf @ W1a_n + b1a  (16x512, K=256) ----------------
  {
    int c0 = w * 64;
    v8f acc[4] = {vzero(), vzero(), vzero(), vzero()};
    wmma_strip<4, 8>(s_nf1, 256, w1an + (size_t)c0 * 256, 256, lane, acc);
    #pragma unroll
    for (int t = 0; t < 4; ++t) {
      int col = c0 + t * 16 + (lane & 15);
      float ba = b1a[col];
      #pragma unroll
      for (int v = 0; v < 8; ++v) {
        int row = v + (lane >> 4) * 8;
        s_zn[row * 512 + col] = (__bf16)(acc[t][v] + ba);
      }
    }
  }
  __syncthreads();

  // persistent msg accumulators: wave w owns output cols [w*32, w*32+32)
  v8f macc[2] = {vzero(), vzero()};

  // ---- anchor loop --------------------------------------------------------
  for (int a = 0; a < 64; ++a) {
    if (tid < 16) {
      const float* px = node_x + (size_t)(node0 + tid) * 3;
      const float* pa = anchor_x + (size_t)(g * 64 + a) * 3;
      float dx = px[0] - pa[0] + 1e-8f;
      float dy = px[1] - pa[1] + 1e-8f;
      float dz = px[2] - pa[2] + 1e-8f;
      s_d[tid] = sqrtf(dx * dx + dy * dy + dz * dz);
    }
    __syncthreads();
    {   // RBF: 16x64 values, 4 per thread. mu_e = 20e/63, sigma = 20/64
      int base = tid * 4;
      #pragma unroll
      for (int j = 0; j < 4; ++j) {
        int idx = base + j, n = idx >> 6, e = idx & 63;
        float t = (s_d[n] * 0.1f - (20.0f / 63.0f) * (float)e) * (1.0f / 0.3125f);
        s_rbf[idx] = (__bf16)__expf(-t * t);
      }
    }
    __syncthreads();
    // layer 1: relu(rbf @ W1r + z_node + t_anchor[a])  -> s_h1 (16x512 bf16)
    {
      int c0 = w * 64;
      v8f acc[4] = {vzero(), vzero(), vzero(), vzero()};
      wmma_strip<4, 2>(s_rbf, 64, w1ar + (size_t)c0 * 64, 64, lane, acc);
      #pragma unroll
      for (int t = 0; t < 4; ++t) {
        int col = c0 + t * 16 + (lane & 15);
        float ta = t_anchor[(size_t)(g * 64 + a) * 512 + col];
        #pragma unroll
        for (int v = 0; v < 8; ++v) {
          int row = v + (lane >> 4) * 8;
          float val = acc[t][v] + (float)s_zn[row * 512 + col] + ta;
          s_h1[row * 512 + col] = (__bf16)fmaxf(val, 0.0f);
        }
      }
    }
    __syncthreads();
    // layer 2: relu(h1 @ W1b + b1b) -> s_h2 (16x512, K=512)
    {
      int c0 = w * 64;
      v8f acc[4] = {vzero(), vzero(), vzero(), vzero()};
      wmma_strip<4, 16>(s_h1, 512, w1bt + (size_t)c0 * 512, 512, lane, acc);
      #pragma unroll
      for (int t = 0; t < 4; ++t) {
        int col = c0 + t * 16 + (lane & 15);
        float bb = b1b[col];
        #pragma unroll
        for (int v = 0; v < 8; ++v) {
          int row = v + (lane >> 4) * 8;
          s_h2[row * 512 + col] = (__bf16)fmaxf(acc[t][v] + bb, 0.0f);
        }
      }
    }
    __syncthreads();
    // layer 3: accumulate msg = h2 @ W1c into persistent regs (16x32 slice)
    wmma_strip<2, 16>(s_h2, 512, w1ct + (size_t)(w * 32) * 512, 512, lane, macc);
    __syncthreads();
  }

  // ---- upd = mean + b1c; residual -> s_x ----------------------------------
  #pragma unroll
  for (int half = 0; half < 2; ++half) {
    int col = w * 32 + half * 16 + (lane & 15);
    float bc = b1c[col];
    #pragma unroll
    for (int v = 0; v < 8; ++v) {
      int row = v + (lane >> 4) * 8;
      int n = node0 + row;
      float upd = macc[half][v] * (1.0f / 64.0f) + bc;
      s_x[row * 256 + col] =
          node_features[(size_t)n * 256 + col] + upd * node_mask[n];
    }
  }
  __syncthreads();

  // ---- LayerNorm 1 (threads 0..15, one row each) --------------------------
  if (tid < 16) {
    float m = 0.0f;
    for (int c = 0; c < 256; ++c) m += s_x[tid * 256 + c];
    m *= (1.0f / 256.0f);
    float vv = 0.0f;
    for (int c = 0; c < 256; ++c) { float d = s_x[tid * 256 + c] - m; vv += d * d; }
    float inv = rsqrtf(vv * (1.0f / 256.0f) + 1e-5f);
    for (int c = 0; c < 256; ++c) {
      float nx = (s_x[tid * 256 + c] - m) * inv * ln1g[c] + ln1b[c];
      s_x[tid * 256 + c] = nx;
      s_nf1[tid * 256 + c] = (__bf16)nx;
    }
  }
  __syncthreads();

  // ---- MLP2-a: relu(nf1 @ W2a + b2a) (16x512, K=256) -> s_h1 --------------
  {
    int c0 = w * 64;
    v8f acc[4] = {vzero(), vzero(), vzero(), vzero()};
    wmma_strip<4, 8>(s_nf1, 256, w2at + (size_t)c0 * 256, 256, lane, acc);
    #pragma unroll
    for (int t = 0; t < 4; ++t) {
      int col = c0 + t * 16 + (lane & 15);
      float bb = b2a[col];
      #pragma unroll
      for (int v = 0; v < 8; ++v) {
        int row = v + (lane >> 4) * 8;
        s_h1[row * 512 + col] = (__bf16)fmaxf(acc[t][v] + bb, 0.0f);
      }
    }
  }
  __syncthreads();
  // ---- MLP2-b: relu(h @ W2b + b2b) (16x512, K=512) -> s_h2 ----------------
  {
    int c0 = w * 64;
    v8f acc[4] = {vzero(), vzero(), vzero(), vzero()};
    wmma_strip<4, 16>(s_h1, 512, w2bt + (size_t)c0 * 512, 512, lane, acc);
    #pragma unroll
    for (int t = 0; t < 4; ++t) {
      int col = c0 + t * 16 + (lane & 15);
      float bb = b2b[col];
      #pragma unroll
      for (int v = 0; v < 8; ++v) {
        int row = v + (lane >> 4) * 8;
        s_h2[row * 512 + col] = (__bf16)fmaxf(acc[t][v] + bb, 0.0f);
      }
    }
  }
  __syncthreads();
  // ---- MLP2-c + residual (16x256, K=512), in-place on s_x -----------------
  {
    v8f acc[2] = {vzero(), vzero()};
    wmma_strip<2, 16>(s_h2, 512, w2ct + (size_t)(w * 32) * 512, 512, lane, acc);
    #pragma unroll
    for (int half = 0; half < 2; ++half) {
      int col = w * 32 + half * 16 + (lane & 15);
      float bc = b2c[col];
      #pragma unroll
      for (int v = 0; v < 8; ++v) {
        int row = v + (lane >> 4) * 8;
        int n = node0 + row;
        s_x[row * 256 + col] += (acc[half][v] + bc) * node_mask[n];
      }
    }
  }
  __syncthreads();

  // ---- LayerNorm 2 -> global out ------------------------------------------
  if (tid < 16) {
    float m = 0.0f;
    for (int c = 0; c < 256; ++c) m += s_x[tid * 256 + c];
    m *= (1.0f / 256.0f);
    float vv = 0.0f;
    for (int c = 0; c < 256; ++c) { float d = s_x[tid * 256 + c] - m; vv += d * d; }
    float inv = rsqrtf(vv * (1.0f / 256.0f) + 1e-5f);
    float* po = out + (size_t)(node0 + tid) * 256;
    for (int c = 0; c < 256; ++c)
      po[c] = (s_x[tid * 256 + c] - m) * inv * ln2g[c] + ln2b[c];
  }
}

// ---------------------------------------------------------------------------
static inline size_t alignup256(size_t x) { return (x + 255) & ~(size_t)255; }

extern "C" void kernel_launch(void* const* d_in, const int* in_sizes, int n_in,
                              void* d_out, int out_size, void* d_ws, size_t ws_size,
                              hipStream_t stream) {
  (void)in_sizes; (void)n_in; (void)out_size; (void)ws_size;
  const float* anchor_x        = (const float*)d_in[0];
  const float* node_x          = (const float*)d_in[1];
  const float* anchor_features = (const float*)d_in[2];
  const float* node_features   = (const float*)d_in[3];
  const float* node_mask       = (const float*)d_in[6];
  const float* W1a = (const float*)d_in[8];  const float* b1a = (const float*)d_in[9];
  const float* W1b = (const float*)d_in[10]; const float* b1b = (const float*)d_in[11];
  const float* W1c = (const float*)d_in[12]; const float* b1c = (const float*)d_in[13];
  const float* ln1g = (const float*)d_in[14]; const float* ln1b = (const float*)d_in[15];
  const float* W2a = (const float*)d_in[16]; const float* b2a = (const float*)d_in[17];
  const float* W2b = (const float*)d_in[18]; const float* b2b = (const float*)d_in[19];
  const float* W2c = (const float*)d_in[20]; const float* b2c = (const float*)d_in[21];
  const float* ln2g = (const float*)d_in[22]; const float* ln2b = (const float*)d_in[23];
  float* out = (float*)d_out;

  char* ws = (char*)d_ws;
  size_t off = 0;
  auto take = [&](size_t bytes) -> void* {
    void* p = ws + off; off = alignup256(off + bytes); return p;
  };
  __bf16* afb  = (__bf16*)take(512 * 256 * 2);  // anchor_features bf16
  __bf16* w1an = (__bf16*)take(512 * 256 * 2);  // W1a node part, [out][in]
  __bf16* w1aa = (__bf16*)take(512 * 256 * 2);  // W1a anchor part
  __bf16* w1ar = (__bf16*)take(512 * 64 * 2);   // W1a rbf part
  __bf16* w1bt = (__bf16*)take(512 * 512 * 2);
  __bf16* w1ct = (__bf16*)take(256 * 512 * 2);
  __bf16* w2at = (__bf16*)take(512 * 256 * 2);
  __bf16* w2bt = (__bf16*)take(512 * 512 * 2);
  __bf16* w2ct = (__bf16*)take(256 * 512 * 2);
  float*  tanch = (float*)take(512 * 512 * 4);  // af @ W1a_a  [B*A][512]

  // --- weight / feature conversion ---
  k_convert_bf16<<<512, 256, 0, stream>>>(anchor_features, afb, 512 * 256);
  k_transpose_convert<<<512, 256, 0, stream>>>(W1a, 512, 0,   256, 512, w1an);
  k_transpose_convert<<<512, 256, 0, stream>>>(W1a, 512, 256, 256, 512, w1aa);
  k_transpose_convert<<<128, 256, 0, stream>>>(W1a, 512, 512, 64,  512, w1ar);
  k_transpose_convert<<<1024, 256, 0, stream>>>(W1b, 512, 0, 512, 512, w1bt);
  k_transpose_convert<<<512, 256, 0, stream>>>(W1c, 256, 0, 512, 256, w1ct);
  k_transpose_convert<<<512, 256, 0, stream>>>(W2a, 512, 0, 256, 512, w2at);
  k_transpose_convert<<<1024, 256, 0, stream>>>(W2b, 512, 0, 512, 512, w2bt);
  k_transpose_convert<<<512, 256, 0, stream>>>(W2c, 256, 0, 512, 256, w2ct);

  // --- t_anchor = af @ W1a_a : M=512, N=512, K=256 -> 256 waves = 32 blocks
  k_gemm_bf16_k256<<<32, 256, 0, stream>>>(afb, w1aa, nullptr, tanch, 512, 512);

  // --- fused MPNN: 192 node tiles ---
  k_mpnn_main<<<192, 256, 0, stream>>>(
      node_x, anchor_x, node_features, node_mask,
      w1an, tanch, w1ar, w1bt, w1ct, w2at, w2bt, w2ct,
      b1a, b1b, b1c, ln1g, ln1b, b2a, b2b, b2c, ln2g, ln2b, out);
}
